// DependencyParser_4492535792544
// MI455X (gfx1250) — matrix operand order repaired and checked by reading, not compiled, not run on hardware
//
#include <hip/hip_runtime.h>
#include <hip/hip_bf16.h>
#include <stdint.h>

// ---------------------------------------------------------------------------
// BiLSTM dependency parser forward for MI455X (gfx1250, wave32, WMMA).
// Effective LSTM shapes (after the reference's reshape quirk):
//   T = 32 (time), N = 128 (batch), H = 400, gates = 1600.
// Row convention for all (4096 x K) activation matrices: r = n*32 + t.
// Gate columns are stored PERMUTED: n' = 4*h + gate  (gate = i,f,g,o),
// so one 128-wide GEMM tile holds 32 complete gate quadruples and the LSTM
// cell update fuses into the GEMM epilogue.
// ---------------------------------------------------------------------------

typedef __attribute__((ext_vector_type(16))) _Float16 v16h;
typedef __attribute__((ext_vector_type(8)))  _Float16 v8h;
typedef __attribute__((ext_vector_type(8)))  float    v8f;

#define LDSS 40      // LDS row stride in halfs (80B: 16B-aligned, bank-skewed)
#define TM   64      // block tile M
#define TN   128     // block tile N
#define GN   1600    // LSTM gate width
#define GK   416     // padded hidden K (400 -> 416)
#define CTLD 132     // fp32 gate-tile LDS stride

__device__ __forceinline__ float sigmf(float x) {
    return 1.0f / (1.0f + __expf(-x));
}

// A-fragment (16x32 f16): lane m = lane&15; kbase = (lane>=16)?8:0;
// halfs 0..7 -> k = kbase..kbase+7 ; halfs 8..15 -> k = kbase+16..kbase+23.
__device__ __forceinline__ v16h frag_a(const _Float16* As, int mLocal, int lane) {
    const _Float16* p = As + mLocal * LDSS + ((lane >> 4) << 3);
    v8h c0 = *(const v8h*)(p);
    v8h c1 = *(const v8h*)(p + 16);
    return __builtin_shufflevector(c0, c1, 0,1,2,3,4,5,6,7,8,9,10,11,12,13,14,15);
}

// B-fragment (32x16 f16, stored N-major): lane n = lane&15;
// kbase = (lane>=16)?8:0; halfs 0..7 -> k = kbase..kbase+7 ; halfs 8..15 -> k = kbase+16..kbase+23.
__device__ __forceinline__ v16h frag_b(const _Float16* Bs, int nLocal, int lane) {
    const _Float16* p = Bs + nLocal * LDSS + ((lane >> 4) << 3);
    v8h c0 = *(const v8h*)(p);
    v8h c1 = *(const v8h*)(p + 16);
    return __builtin_shufflevector(c0, c1, 0,1,2,3,4,5,6,7,8,9,10,11,12,13,14,15);
}

// ---------------------------------------------------------------------------
// Generic NT WMMA GEMM: C[M,N] = A[M,K](f16) * W[N,K](f16)^T (+bias0/1[n])
// K % 32 == 0, M % 64 == 0. Branch-free inner loop: OOB rows are CLAMPED
// (junk only reaches never-stored outputs). LDS double-buffered.
// ---------------------------------------------------------------------------
__global__ __launch_bounds__(256)
void k_wmma_gemm(const _Float16* __restrict__ A, int lda,
                 const _Float16* __restrict__ W, int ldw,
                 float* __restrict__ C, int ldc,
                 const float* __restrict__ bias0,
                 const float* __restrict__ bias1,
                 int M, int N, int K)
{
    __shared__ _Float16 As[2][TM * LDSS];
    __shared__ _Float16 Bs[2][TN * LDSS];

    const int tid  = threadIdx.x;
    const int lane = tid & 31;
    const int wave = tid >> 5;
    const int wm   = wave & 1;
    const int wn   = wave >> 1;
    const int blockM = blockIdx.y * TM;
    const int blockN = blockIdx.x * TN;

    const v8f zacc = {0.f,0.f,0.f,0.f,0.f,0.f,0.f,0.f};
    v8f acc[2][2] = {{zacc, zacc}, {zacc, zacc}};

    const int arow = tid >> 2;            // 0..63
    const int acol = (tid & 3) << 3;      // 0,8,16,24
    const int brow = tid >> 1;            // 0..127
    const int bcol = (tid & 1) << 4;      // 0,16

    const _Float16* ap = A + (size_t)min(blockM + arow, M - 1) * lda + acol;
    const _Float16* wp = W + (size_t)min(blockN + brow, N - 1) * ldw + bcol;

    // prologue: stage tile 0
    {
        v8h av  = *(const v8h*)(ap);
        v8h bv0 = *(const v8h*)(wp);
        v8h bv1 = *(const v8h*)(wp + 8);
        *(v8h*)(&As[0][arow * LDSS + acol])     = av;
        *(v8h*)(&Bs[0][brow * LDSS + bcol])     = bv0;
        *(v8h*)(&Bs[0][brow * LDSS + bcol + 8]) = bv1;
    }
    __syncthreads();

    int buf = 0;
    for (int k0 = 0; k0 < K; k0 += 32, buf ^= 1) {
        const bool more = (k0 + 32) < K;
        v8h av, bv0, bv1;
        if (more) {                                    // issue next-tile loads
            av  = *(const v8h*)(ap + k0 + 32);
            bv0 = *(const v8h*)(wp + k0 + 32);
            bv1 = *(const v8h*)(wp + k0 + 40);
            __builtin_prefetch(ap + k0 + 64, 0, 1);    // global_prefetch_b8
            __builtin_prefetch(wp + k0 + 64, 0, 1);
        }

        v16h af[2], bf[2];
        #pragma unroll
        for (int mi = 0; mi < 2; ++mi)
            af[mi] = frag_a(As[buf], wm * 32 + mi * 16 + (lane & 15), lane);
        #pragma unroll
        for (int ni = 0; ni < 2; ++ni)
            bf[ni] = frag_b(Bs[buf], wn * 32 + ni * 16 + (lane & 15), lane);
        #pragma unroll
        for (int mi = 0; mi < 2; ++mi)
            #pragma unroll
            for (int ni = 0; ni < 2; ++ni)
                acc[mi][ni] = __builtin_amdgcn_wmma_f32_16x16x32_f16(
                    false, af[mi], false, bf[ni], (short)0, acc[mi][ni],
                    false, false);

        if (more) {
            *(v8h*)(&As[buf ^ 1][arow * LDSS + acol])     = av;
            *(v8h*)(&Bs[buf ^ 1][brow * LDSS + bcol])     = bv0;
            *(v8h*)(&Bs[buf ^ 1][brow * LDSS + bcol + 8]) = bv1;
        }
        __syncthreads();
    }

    const int mOff = (lane >> 4) << 3;
    const int nOff = lane & 15;
    #pragma unroll
    for (int mi = 0; mi < 2; ++mi)
        #pragma unroll
        for (int ni = 0; ni < 2; ++ni)
            #pragma unroll
            for (int v = 0; v < 8; ++v) {
                int row = blockM + wm * 32 + mi * 16 + mOff + v;
                int col = blockN + wn * 32 + ni * 16 + nOff;
                if (col < N) {
                    float r = acc[mi][ni][v];
                    if (bias0) r += bias0[col];
                    if (bias1) r += bias1[col];
                    C[(size_t)row * ldc + col] = r;
                }
            }
}

// ---------------------------------------------------------------------------
// Fused LSTM step: g = h_prev * Whh^T + xg[t]  (gate-permuted columns),
// then the cell update in the same kernel via an LDS fp32 tile.
//   grid = (ceil(GN/TN)=13, 128/TM=2, 2 dirs), block = 256.
// Hin/Hout are parity double-buffers (readers of old h vs writers of new h
// are different blocks, so the state cannot be updated in place).
// ---------------------------------------------------------------------------
__global__ __launch_bounds__(256)
void k_lstm_step(const _Float16* __restrict__ Hin,   // [2dir][128][416]
                 const _Float16* __restrict__ Whh,   // [2dir][1600][416] perm rows
                 const float* __restrict__ XG,       // [2dir][4096][1600] perm cols
                 float* __restrict__ Cst,            // [2dir][128][400]
                 _Float16* __restrict__ Hout,        // [2dir][128][416]
                 _Float16* __restrict__ Eout, int ldE,
                 int s)
{
    __shared__ union SM {
        struct { _Float16 a[2][TM * LDSS]; _Float16 b[2][TN * LDSS]; } st;
        float ct[TM * CTLD];
    } sm;

    const int tid  = threadIdx.x;
    const int lane = tid & 31;
    const int wave = tid >> 5;
    const int wm   = wave & 1;
    const int wn   = wave >> 1;
    const int dir  = blockIdx.z;
    const int blockM = blockIdx.y * TM;          // batch rows
    const int blockN = blockIdx.x * TN;          // permuted gate cols
    const int t = dir ? (31 - s) : s;

    const _Float16* A = Hin + (size_t)dir * 128 * GK;
    const _Float16* W = Whh + (size_t)dir * GN * GK;
    const float* xg   = XG  + (size_t)dir * 4096 * GN + (size_t)t * GN;

    const v8f zacc = {0.f,0.f,0.f,0.f,0.f,0.f,0.f,0.f};
    v8f acc[2][2] = {{zacc, zacc}, {zacc, zacc}};

    const int arow = tid >> 2;
    const int acol = (tid & 3) << 3;
    const int brow = tid >> 1;
    const int bcol = (tid & 1) << 4;

    const _Float16* ap = A + (size_t)(blockM + arow) * GK + acol;
    const _Float16* wp = W + (size_t)min(blockN + brow, GN - 1) * GK + bcol;

    {
        v8h av  = *(const v8h*)(ap);
        v8h bv0 = *(const v8h*)(wp);
        v8h bv1 = *(const v8h*)(wp + 8);
        *(v8h*)(&sm.st.a[0][arow * LDSS + acol])     = av;
        *(v8h*)(&sm.st.b[0][brow * LDSS + bcol])     = bv0;
        *(v8h*)(&sm.st.b[0][brow * LDSS + bcol + 8]) = bv1;
    }
    __syncthreads();

    int buf = 0;
    for (int k0 = 0; k0 < GK; k0 += 32, buf ^= 1) {
        const bool more = (k0 + 32) < GK;
        v8h av, bv0, bv1;
        if (more) {
            av  = *(const v8h*)(ap + k0 + 32);
            bv0 = *(const v8h*)(wp + k0 + 32);
            bv1 = *(const v8h*)(wp + k0 + 40);
            __builtin_prefetch(wp + k0 + 64, 0, 1);
        }
        v16h af[2], bf[2];
        #pragma unroll
        for (int mi = 0; mi < 2; ++mi)
            af[mi] = frag_a(sm.st.a[buf], wm * 32 + mi * 16 + (lane & 15), lane);
        #pragma unroll
        for (int ni = 0; ni < 2; ++ni)
            bf[ni] = frag_b(sm.st.b[buf], wn * 32 + ni * 16 + (lane & 15), lane);
        #pragma unroll
        for (int mi = 0; mi < 2; ++mi)
            #pragma unroll
            for (int ni = 0; ni < 2; ++ni)
                acc[mi][ni] = __builtin_amdgcn_wmma_f32_16x16x32_f16(
                    false, af[mi], false, bf[ni], (short)0, acc[mi][ni],
                    false, false);
        if (more) {
            *(v8h*)(&sm.st.a[buf ^ 1][arow * LDSS + acol])     = av;
            *(v8h*)(&sm.st.b[buf ^ 1][brow * LDSS + bcol])     = bv0;
            *(v8h*)(&sm.st.b[buf ^ 1][brow * LDSS + bcol + 8]) = bv1;
        }
        __syncthreads();
    }

    // dump g = acc + xg into the fp32 LDS tile
    const int mOff = (lane >> 4) << 3;
    const int nOff = lane & 15;
    #pragma unroll
    for (int mi = 0; mi < 2; ++mi)
        #pragma unroll
        for (int ni = 0; ni < 2; ++ni)
            #pragma unroll
            for (int v = 0; v < 8; ++v) {
                int rl  = wm * 32 + mi * 16 + mOff + v;      // local batch row
                int cl  = wn * 32 + ni * 16 + nOff;          // local gate col
                int col = blockN + cl;
                float r = acc[mi][ni][v];
                if (col < GN) r += xg[(size_t)(blockM + rl) * (32 * GN) + col];
                sm.ct[rl * CTLD + cl] = r;
            }
    __syncthreads();

    // cell update: 64 rows x 32 h-quads per block, 8 elements per thread
    const int hBase = blockN >> 2;
    #pragma unroll
    for (int e = 0; e < 8; ++e) {
        int idx = tid + e * 256;          // 0..2047
        int nl  = idx >> 5;               // 0..63
        int hl  = idx & 31;               // 0..31
        int h   = hBase + hl;
        if (h < 400) {
            int n = blockM + nl;
            const float* g = &sm.ct[nl * CTLD + hl * 4];
            float gi = g[0], gf = g[1], gg = g[2], go = g[3];
            float* cp = Cst + (size_t)dir * 128 * 400 + (size_t)n * 400 + h;
            float c  = sigmf(gf) * (*cp) + sigmf(gi) * tanhf(gg);
            *cp = c;
            float hv = sigmf(go) * tanhf(c);
            _Float16 h16 = (_Float16)hv;
            Hout[(size_t)dir * 128 * GK + (size_t)n * GK + h] = h16;
            Eout[(size_t)(n * 32 + t) * ldE + dir * 400 + h]  = h16;
        }
    }
}

// ---------------------------------------------------------------------------
// Embedding gather -> fp16 E0[4096][416] (cols 400..415 = 0); r: b=r>>7,l=r&127
// ---------------------------------------------------------------------------
__global__ void k_embed(const int* __restrict__ words, const int* __restrict__ tags,
                        const float* __restrict__ wemb, const float* __restrict__ temb,
                        _Float16* __restrict__ E0)
{
    int r = blockIdx.x, c = threadIdx.x;
    int b = r >> 7, l = r & 127;
    float v = 0.0f;
    if (c < 300)      v = wemb[(size_t)words[b * 128 + l] * 300 + c];
    else if (c < 400) v = temb[(size_t)tags[b * 128 + l] * 100 + (c - 300)];
    E0[(size_t)r * 416 + c] = (_Float16)v;
}

// fp32 -> fp16 weight convert, K-padded, optional gate-permuted row order
// (dst row n takes src row (n&3)*400 + (n>>2)).
__global__ void k_cvt(const float* __restrict__ src, int srcLd, int colOff,
                      _Float16* __restrict__ dst, int rows, int cols, int dstLd,
                      int permute)
{
    int i = blockIdx.x * blockDim.x + threadIdx.x;
    if (i >= rows * dstLd) return;
    int r = i / dstLd, c = i - r * dstLd;
    int sr = permute ? ((r & 3) * 400 + (r >> 2)) : r;
    dst[i] = (c < cols) ? (_Float16)src[(size_t)sr * srcLd + colOff + c]
                        : (_Float16)0.0f;
}

// permuted bias sum: pb[n] = b_ih[p] + b_hh[p], p = (n&3)*400 + (n>>2)
__global__ void k_bias_perm(const float* __restrict__ bih,
                            const float* __restrict__ bhh,
                            float* __restrict__ pb)
{
    int n = blockIdx.x * blockDim.x + threadIdx.x;
    if (n < GN) {
        int p = (n & 3) * 400 + (n >> 2);
        pb[n] = bih[p] + bhh[p];
    }
}

__global__ void k_zero32(uint32_t* __restrict__ p, int n)
{
    int i = blockIdx.x * blockDim.x + threadIdx.x;
    if (i < n) p[i] = 0u;
}

// ---------------------------------------------------------------------------
// Pairwise scorer: out[(i*128+j)*32+b] = b2 + sum_h w2[h]*tanh(a_ib[h]+bm_jb[h])
// ---------------------------------------------------------------------------
__global__ __launch_bounds__(128)
void k_pair(const float* __restrict__ Aarr, const float* __restrict__ Bm,
            const float* __restrict__ w2, const float* __restrict__ b2,
            float* __restrict__ out)
{
    __shared__ float sa[100];
    __shared__ float sw[100];
    int i = blockIdx.x, b = blockIdx.y, j = threadIdx.x;
    if (threadIdx.x < 100) {
        sa[threadIdx.x] = Aarr[(size_t)(i * 32 + b) * 100 + threadIdx.x];
        sw[threadIdx.x] = w2[threadIdx.x];
    }
    __syncthreads();
    const float* bp = Bm + (size_t)(j * 32 + b) * 100;
    float s = 0.0f;
    #pragma unroll 4
    for (int h = 0; h < 100; ++h)
        s += sw[h] * tanhf(sa[h] + bp[h]);
    out[((size_t)(i * 128 + j)) * 32 + b] = s + b2[0];
}

// ---------------------------------------------------------------------------

static inline int cdiv(int a, int b) { return (a + b - 1) / b; }

extern "C" void kernel_launch(void* const* d_in, const int* in_sizes, int n_in,
                              void* d_out, int out_size, void* d_ws, size_t ws_size,
                              hipStream_t stream)
{
    (void)in_sizes; (void)n_in; (void)out_size; (void)ws_size;

    const int*   words = (const int*)d_in[0];
    const int*   tags  = (const int*)d_in[1];
    const float* wemb  = (const float*)d_in[4];
    const float* temb  = (const float*)d_in[5];
    const float* w_ih[4] = {(const float*)d_in[6],  (const float*)d_in[10],
                            (const float*)d_in[14], (const float*)d_in[18]};
    const float* w_hh[4] = {(const float*)d_in[7],  (const float*)d_in[11],
                            (const float*)d_in[15], (const float*)d_in[19]};
    const float* b_ih[4] = {(const float*)d_in[8],  (const float*)d_in[12],
                            (const float*)d_in[16], (const float*)d_in[20]};
    const float* b_hh[4] = {(const float*)d_in[9],  (const float*)d_in[13],
                            (const float*)d_in[17], (const float*)d_in[21]};
    const float* w1 = (const float*)d_in[22];
    const float* b1 = (const float*)d_in[23];
    const float* w2 = (const float*)d_in[24];
    const float* b2 = (const float*)d_in[25];
    float* out = (float*)d_out;

    // ---- workspace carve-up (256B aligned) ----
    char* base = (char*)d_ws;
    size_t off = 0;
    auto carve = [&](size_t bytes) -> char* {
        char* p = base + off;
        off = (off + bytes + 255) & ~(size_t)255;
        return p;
    };
    _Float16* E0   = (_Float16*)carve((size_t)4096 * 416 * 2);
    _Float16* E1   = (_Float16*)carve((size_t)4096 * 832 * 2);
    _Float16* E2   = (_Float16*)carve((size_t)4096 * 832 * 2);
    _Float16* WihA = (_Float16*)carve((size_t)GN * 832 * 2);      // dir0 / W1a
    _Float16* WihB = (_Float16*)carve((size_t)GN * 832 * 2);      // dir1 / W1b
    _Float16* Whh  = (_Float16*)carve((size_t)2 * GN * GK * 2);   // both dirs
    float*    XG   = (float*)carve((size_t)2 * 4096 * GN * 4);
    _Float16* Hst  = (_Float16*)carve((size_t)2 * 2 * 128 * GK * 2); // parity x dir
    float*    Cst  = (float*)carve((size_t)2 * 128 * 400 * 4);
    float*    Aout = (float*)carve((size_t)4096 * 100 * 4);
    float*    Bout = (float*)carve((size_t)4096 * 100 * 4);
    float*    PB   = (float*)carve((size_t)2 * GN * 4);

    const size_t HPAR = (size_t)2 * 128 * GK;   // one parity (both dirs), halfs

    // ---- embeddings -> E0 ----
    k_embed<<<4096, 416, 0, stream>>>(words, tags, wemb, temb, E0);

    // ---- two BiLSTM layers ----
    const _Float16* Ein  = E0;
    _Float16*       Eout = E1;
    for (int layer = 0; layer < 2; ++layer) {
        const int d0  = layer * 2;
        const int d1  = layer * 2 + 1;
        const int Din = (layer == 0) ? 400 : 800;
        const int Kin = (layer == 0) ? 416 : 832;

        // weights -> f16, (N x K), K zero-padded, gate-permuted rows
        {
            int nIh = GN * Kin, nHh = GN * GK;
            k_cvt<<<cdiv(nIh, 256), 256, 0, stream>>>(w_ih[d0], Din, 0, WihA, GN, Din, Kin, 1);
            k_cvt<<<cdiv(nIh, 256), 256, 0, stream>>>(w_ih[d1], Din, 0, WihB, GN, Din, Kin, 1);
            k_cvt<<<cdiv(nHh, 256), 256, 0, stream>>>(w_hh[d0], 400, 0, Whh,               GN, 400, GK, 1);
            k_cvt<<<cdiv(nHh, 256), 256, 0, stream>>>(w_hh[d1], 400, 0, Whh + (size_t)GN * GK, GN, 400, GK, 1);
            k_bias_perm<<<cdiv(GN, 256), 256, 0, stream>>>(b_ih[d0], b_hh[d0], PB);
            k_bias_perm<<<cdiv(GN, 256), 256, 0, stream>>>(b_ih[d1], b_hh[d1], PB + GN);
        }

        // xg = Ein * w_ih^T + (b_ih + b_hh)   (M=4096, N=1600, K=Kin)
        {
            dim3 grid(cdiv(GN, TN), cdiv(4096, TM));
            k_wmma_gemm<<<grid, 256, 0, stream>>>(Ein, Kin, WihA, Kin,
                XG, GN, PB, nullptr, 4096, GN, Kin);
            k_wmma_gemm<<<grid, 256, 0, stream>>>(Ein, Kin, WihB, Kin,
                XG + (size_t)4096 * GN, GN, PB + GN, nullptr, 4096, GN, Kin);
        }

        // zero h (both parities incl. K-pad), c, and the next-layer input
        k_zero32<<<cdiv((int)(2 * HPAR / 2), 256), 256, 0, stream>>>((uint32_t*)Hst, (int)(2 * HPAR / 2));
        k_zero32<<<cdiv(2 * 128 * 400, 256), 256, 0, stream>>>((uint32_t*)Cst, 2 * 128 * 400);
        k_zero32<<<cdiv(4096 * 832 / 2, 256), 256, 0, stream>>>((uint32_t*)Eout, 4096 * 832 / 2);

        // recurrent scan: one fused launch per step, both dirs via gridDim.z
        for (int s = 0; s < 32; ++s) {
            _Float16* Hin  = Hst + (size_t)(s & 1) * HPAR;
            _Float16* Hnew = Hst + (size_t)((s + 1) & 1) * HPAR;
            dim3 grid(cdiv(GN, TN), 128 / TM, 2);
            k_lstm_step<<<grid, 256, 0, stream>>>(Hin, Whh, XG, Cst, Hnew,
                                                  Eout, 832, s);
        }

        Ein  = Eout;
        Eout = E2;
    }

    // ---- MLP head: a = E2 * W1a^T ; bm = E2 * W1b^T + b1 ----
    {
        int nW = 100 * 832;
        k_cvt<<<cdiv(nW, 256), 256, 0, stream>>>(w1, 1600, 0,   WihA, 100, 800, 832, 0);
        k_cvt<<<cdiv(nW, 256), 256, 0, stream>>>(w1, 1600, 800, WihB, 100, 800, 832, 0);
        dim3 grid(cdiv(100, TN), cdiv(4096, TM));
        k_wmma_gemm<<<grid, 256, 0, stream>>>((const _Float16*)E2, 832, WihA, 832,
            Aout, 100, nullptr, nullptr, 4096, 100, 832);
        k_wmma_gemm<<<grid, 256, 0, stream>>>((const _Float16*)E2, 832, WihB, 832,
            Bout, 100, b1, nullptr, 4096, 100, 832);
    }

    // ---- pairwise tanh scorer -> d_out (128*128*32 floats) ----
    {
        dim3 grid(128, 32);
        k_pair<<<grid, 128, 0, stream>>>(Aout, Bout, w2, b2, out);
    }
}
